// FSARNNIntegrateEmptyStateSaperateGRU_80513456931341
// MI455X (gfx1250) — matrix operand order repaired and checked by reading, not compiled, not run on hardware
//
#include <hip/hip_runtime.h>
#include <hip/hip_bf16.h>

// ---------------------------------------------------------------------------
// FSA-RNN forward for MI455X (gfx1250, wave32, WMMA).
//
// Sizes: V=50000 D=300 R=256 SA=256 B=512 L=128.
// All GEMMs are mapped to v_wmma_f32_16x16x32_f16 (f16 A/B, f32 accum).
// Weights + Lseq are repacked once into WMMA operand layouts in d_ws:
//   A-operand 16x32 f16 tile (512 halves): lane l, half h ->
//       M = l%16,  K = h + 8*(l/16) + (h>=8 ? 8 : 0)
//   B-operand 32x16 f16 tile (512 halves): lane l, half h ->
//       N = l%16,  K = h + 8*(l/16) + (h>=8 ? 8 : 0)
// so each lane's v16h operand is one contiguous 32B load.
//
// Round-2 changes vs round-1:
//  * K-loops load each A fragment ONCE and feed both of the wave's N-tile
//    accumulators (halves LDS/global A traffic in the scan).
//  * Branch-free sigmoid/tanh on v_exp_f32/v_rcp_f32 (no exec-mask branching
//    interleaved with the WMMA chains, far fewer valu/salu).
//  * global_prefetch of the next timestep's Lt A-pack at the top of each step.
// ---------------------------------------------------------------------------

typedef __attribute__((ext_vector_type(16))) _Float16 v16h;
typedef __attribute__((ext_vector_type(8)))  float    v8f;

#define SA   256
#define RR   256
#define BB   512
#define LL   128
#define KT   8          // 256 / 32 K-tiles for the recurrence GEMMs
#define KT1  10         // 320 / 32 K-tiles for the embedding GEMM (300 padded)
#define WMAT 65536      // halves per packed 256x256 weight matrix
#define GPAD (320 * 256)

#define WMMA(A, B, C) __builtin_amdgcn_wmma_f32_16x16x32_f16( \
    false, (A), false, (B), (short)0, (C), false, false)

// offset of element (m, k) inside A-operand packed stream of 16 x (32*KT)
__device__ __forceinline__ int apack_off(int m, int k) {
  int kt = k >> 5, kk = k & 31;
  int g  = (kk >> 3) & 1;                    // lane group (0: lanes 0-15, 1: 16-31)
  int h  = (kk & 7) + ((kk >> 4) << 3);      // half index 0..15
  return kt * 512 + (m + 16 * g) * 16 + h;
}

// offset of element (k, n) inside B-operand packed 256(or 320) x 256 matrix
__device__ __forceinline__ int bpack_off(int k, int n) {
  int kt = k >> 5, kk = k & 31;
  int nt = n >> 4, nn = n & 15;
  return (kt * 16 + nt) * 512 + (nn + 16 * ((kk >> 3) & 1)) * 16 + (kk & 7) + ((kk >> 4) << 3);
}

__device__ __forceinline__ float fast_rcp(float x) {
#if __has_builtin(__builtin_amdgcn_rcpf)
  return __builtin_amdgcn_rcpf(x);
#else
  return 1.0f / x;
#endif
}

// branch-free sigmoid / tanh: one v_exp_f32 + one v_rcp_f32 each
__device__ __forceinline__ float sigm(float x) {
  return fast_rcp(1.0f + __expf(-x));
}
__device__ __forceinline__ float fast_tanh(float x) {
  float e = __expf(2.0f * x);
  return (e - 1.0f) * fast_rcp(e + 1.0f);
}

__device__ __forceinline__ v8f splat8(float x) {
  v8f v = {x, x, x, x, x, x, x, x};
  return v;
}

__device__ __forceinline__ v16h ldB(const _Float16* __restrict__ W, int kt, int nt, int lane) {
  return *(const v16h*)(W + (kt * 16 + nt) * 512 + lane * 16);
}

// ---------------------------------------------------------------------------
// Kernel 1: repack the 7 recurrence weights (+ embed_r_gen, K padded 300->320)
// from f32 row-major to f16 WMMA B-operand tile layout.
// mats: 0 Wss1, 1 Wrs1, 2 Wss2, 3 Wrs2, 4 trans_r1, 5 trans_r2^T, 6 wildcard
// ---------------------------------------------------------------------------
__global__ void pack_weights_kernel(const float* __restrict__ Wss1,
                                    const float* __restrict__ Wrs1,
                                    const float* __restrict__ Wss2,
                                    const float* __restrict__ Wrs2,
                                    const float* __restrict__ tr1,
                                    const float* __restrict__ tr2,
                                    const float* __restrict__ twc,
                                    const float* __restrict__ G,
                                    _Float16* __restrict__ wp) {
  int i = blockIdx.x * blockDim.x + threadIdx.x;
  const int NW = 7 * WMAT;
  if (i < NW) {
    int mat = i >> 16;
    int e   = i & 65535;
    int k = e >> 8, n = e & 255;
    float v;
    switch (mat) {
      case 0:  v = Wss1[k * SA + n]; break;
      case 1:  v = Wrs1[k * SA + n]; break;
      case 2:  v = Wss2[k * SA + n]; break;
      case 3:  v = Wrs2[k * SA + n]; break;
      case 4:  v = tr1[k * RR + n];  break;
      case 5:  v = tr2[n * RR + k];  break;   // used as trans_r2^T: K=R, N=SA
      default: v = twc[k * SA + n];  break;
    }
    wp[mat * WMAT + bpack_off(k, n)] = (_Float16)v;
  } else {
    int j = i - NW;
    if (j < GPAD) {                           // embed_r_gen, K in [0,320)
      int k = j >> 8, n = j & 255;
      float v = (k < 300) ? G[k * RR + n] : 0.0f;
      wp[NW + bpack_off(k, n)] = (_Float16)v;
    }
  }
}

// ---------------------------------------------------------------------------
// Kernel 2: per-token language signal
//   Lg = tanh(emb @ embed_r_gen);  Lseq = regex*beta + Lg*(1-beta)
// Grid (32, 128): block (bx, t) handles batch rows [16*bx, 16*bx+16) at step t.
// Writes Lseq in f16 A-operand layout: Lp[(bx*128+t)*4096 + apack_off(m, r)].
// ---------------------------------------------------------------------------
__global__ void phase1_kernel(const int*   __restrict__ ids,
                              const float* __restrict__ embedding,
                              const float* __restrict__ embed_r,
                              const float* __restrict__ beta,
                              const _Float16* __restrict__ Gp,
                              _Float16* __restrict__ Lp) {
  __shared__ _Float16 eA[KT1 * 512];   // 16 x 320 f16 A-pack of embeddings
  __shared__ int sid[16];

  const int tid = threadIdx.x;
  const int b0  = blockIdx.x * 16;
  const int t   = blockIdx.y;

  if (tid < 16) sid[tid] = ids[(b0 + tid) * LL + t];
  __syncthreads();

  // gather 16 embedding rows (K padded to 320 with zeros) into the A-pack
  for (int e = 0; e < 20; ++e) {
    int idx = tid + 256 * e;              // 5120 = 16 * 320 elements
    int m = idx / 320, k = idx - m * 320;
    float v = (k < 300) ? embedding[(long)sid[m] * 300 + k] : 0.0f;
    eA[apack_off(m, k)] = (_Float16)v;
  }
  __syncthreads();

  const int wave = tid >> 5, lane = tid & 31;
  const long lbase = (long)(blockIdx.x * LL + t) * (KT * 512);
  const int nt0 = wave * 2, nt1 = nt0 + 1;

  v8f acc0 = {}, acc1 = {};
#pragma unroll
  for (int kt = 0; kt < KT1; ++kt) {
    v16h a = *(const v16h*)(eA + kt * 512 + lane * 16);   // shared A fragment
    acc0 = WMMA(a, ldB(Gp, kt, nt0, lane), acc0);
    acc1 = WMMA(a, ldB(Gp, kt, nt1, lane), acc1);
  }

  const int m_hi = (lane >> 4) << 3;
  const int r0 = nt0 * 16 + (lane & 15);
  const int r1 = nt1 * 16 + (lane & 15);
  const float bv0 = beta[r0], bv1 = beta[r1];
#pragma unroll
  for (int j = 0; j < 8; ++j) {
    int m = j + m_hi;
    long erow = (long)sid[m] * RR;
    float v0 = embed_r[erow + r0] * bv0 + fast_tanh(acc0[j]) * (1.0f - bv0);
    float v1 = embed_r[erow + r1] * bv1 + fast_tanh(acc1[j]) * (1.0f - bv1);
    Lp[lbase + apack_off(m, r0)] = (_Float16)v0;
    Lp[lbase + apack_off(m, r1)] = (_Float16)v1;
  }
}

// ---------------------------------------------------------------------------
// Kernel 3: the GRU-style scan. 32 workgroups x 256 threads (8 waves).
// Each workgroup owns 16 batch rows; hidden state lives in LDS for all 128
// steps. Each wave owns two 16-wide N-tiles of the 256-wide state.
// Per step: 112 v_wmma_f32_16x16x32_f16 per wave (896 per workgroup); each A fragment is
// loaded once per K-tile and feeds both N-tile accumulators.
// ---------------------------------------------------------------------------
__global__ void phase2_kernel(const _Float16* __restrict__ Wp,
                              const _Float16* __restrict__ Lp,
                              const float* __restrict__ bs1,
                              const float* __restrict__ bs2,
                              const float* __restrict__ h1,
                              float* __restrict__ out) {
  extern __shared__ char smem_raw[];
  _Float16* hA  = (_Float16*)smem_raw;      // hidden, A-pack   (4096 halves)
  _Float16* hbA = hA  + 4096;               // hbar,   A-pack
  _Float16* pA  = hbA + 4096;               // Lt*Rv,  A-pack
  float* hf  = (float*)(pA + 4096);         // hidden f32       (16 x 256)
  float* zt  = hf  + 4096;                  // sigmoid(z) gate
  float* tmp = zt  + 4096;                  // r-gate -> Rv -> hid (reused)

  const _Float16* Wss1p = Wp;
  const _Float16* Wrs1p = Wp + 1 * WMAT;
  const _Float16* Wss2p = Wp + 2 * WMAT;
  const _Float16* Wrs2p = Wp + 3 * WMAT;
  const _Float16* tr1p  = Wp + 4 * WMAT;
  const _Float16* tr2tp = Wp + 5 * WMAT;
  const _Float16* twcp  = Wp + 6 * WMAT;

  const int tid  = threadIdx.x;
  const int wave = tid >> 5, lane = tid & 31;
  const int b0   = blockIdx.x * 16;
  const int n_lo = lane & 15;
  const int m_hi = (lane >> 4) << 3;
  const int nt0  = wave * 2, nt1 = nt0 + 1;
  const int n0   = nt0 * 16 + n_lo;
  const int n1   = nt1 * 16 + n_lo;

  // h0: one-hot state 0
  for (int e = 0; e < 16; ++e) {
    int idx = tid + 256 * e;
    int m = idx >> 8, n = idx & 255;
    float v = (n == 0) ? 1.0f : 0.0f;
    hf[idx] = v;
    hA[apack_off(m, n)] = (_Float16)v;
  }
  __syncthreads();

  for (int t = 0; t < LL; ++t) {
    const _Float16* lA = Lp + (long)(blockIdx.x * LL + t) * (KT * 512);

    // prefetch next step's Lt A-pack (8KB spread over the 256 threads)
    if (t + 1 < LL) {
      const _Float16* lN = Lp + (long)(blockIdx.x * LL + t + 1) * (KT * 512);
      __builtin_prefetch(lN + tid * 16, 0, 0);
    }

    // ---- gates: z = sig(h@Wss1 + Lt@Wrs1 + bs1), r = sig(h@Wss2 + Lt@Wrs2 + bs2)
    {
      v8f az0 = splat8(bs1[n0]), az1 = splat8(bs1[n1]);
      v8f ar0 = splat8(bs2[n0]), ar1 = splat8(bs2[n1]);
#pragma unroll
      for (int kt = 0; kt < KT; ++kt) {
        v16h ah = *(const v16h*)(hA + kt * 512 + lane * 16);
        v16h al = *(const v16h*)(lA + kt * 512 + lane * 16);
        az0 = WMMA(ah, ldB(Wss1p, kt, nt0, lane), az0);
        az0 = WMMA(al, ldB(Wrs1p, kt, nt0, lane), az0);
        ar0 = WMMA(ah, ldB(Wss2p, kt, nt0, lane), ar0);
        ar0 = WMMA(al, ldB(Wrs2p, kt, nt0, lane), ar0);
        az1 = WMMA(ah, ldB(Wss1p, kt, nt1, lane), az1);
        az1 = WMMA(al, ldB(Wrs1p, kt, nt1, lane), az1);
        ar1 = WMMA(ah, ldB(Wss2p, kt, nt1, lane), ar1);
        ar1 = WMMA(al, ldB(Wrs2p, kt, nt1, lane), ar1);
      }
#pragma unroll
      for (int j = 0; j < 8; ++j) {
        int row = (j + m_hi) * 256;
        zt[row + n0]  = sigm(az0[j]);
        zt[row + n1]  = sigm(az1[j]);
        tmp[row + n0] = sigm(ar0[j]);
        tmp[row + n1] = sigm(ar1[j]);
      }
    }
    __syncthreads();

    // ---- hbar = (1-r)*h1 + r*h  (packed to f16 A-operand)
    for (int e = 0; e < 16; ++e) {
      int idx = tid + 256 * e;
      int m = idx >> 8, n = idx & 255;
      float r  = tmp[idx];
      float hb = (1.0f - r) * h1[n] + r * hf[idx];
      hbA[apack_off(m, n)] = (_Float16)hb;
    }
    __syncthreads();

    // ---- Rv = hbar @ trans_r1
    {
      v8f acc0 = {}, acc1 = {};
#pragma unroll
      for (int kt = 0; kt < KT; ++kt) {
        v16h a = *(const v16h*)(hbA + kt * 512 + lane * 16);
        acc0 = WMMA(a, ldB(tr1p, kt, nt0, lane), acc0);
        acc1 = WMMA(a, ldB(tr1p, kt, nt1, lane), acc1);
      }
#pragma unroll
      for (int j = 0; j < 8; ++j) {
        int row = (j + m_hi) * 256;
        tmp[row + n0] = acc0[j];
        tmp[row + n1] = acc1[j];
      }
    }
    __syncthreads();

    // ---- P = Lt * Rv  (reread Lt from its packed layout, repack product)
    for (int e = 0; e < 16; ++e) {
      int idx = tid + 256 * e;
      int m = idx >> 8, r = idx & 255;
      int po = apack_off(m, r);
      float lt = (float)lA[po];
      pA[po] = (_Float16)(lt * tmp[idx]);
    }
    __syncthreads();

    // ---- hid = relu(P @ trans_r2^T + hbar @ wildcard)
    {
      v8f acc0 = {}, acc1 = {};
#pragma unroll
      for (int kt = 0; kt < KT; ++kt) {
        v16h ap = *(const v16h*)(pA  + kt * 512 + lane * 16);
        v16h ab = *(const v16h*)(hbA + kt * 512 + lane * 16);
        acc0 = WMMA(ap, ldB(tr2tp, kt, nt0, lane), acc0);
        acc0 = WMMA(ab, ldB(twcp,  kt, nt0, lane), acc0);
        acc1 = WMMA(ap, ldB(tr2tp, kt, nt1, lane), acc1);
        acc1 = WMMA(ab, ldB(twcp,  kt, nt1, lane), acc1);
      }
#pragma unroll
      for (int j = 0; j < 8; ++j) {
        int row = (j + m_hi) * 256;
        tmp[row + n0] = fmaxf(acc0[j], 0.0f);
        tmp[row + n1] = fmaxf(acc1[j], 0.0f);
      }
    }
    __syncthreads();

    // ---- h_{t+1} = (1-z)*h + z*hid ; emit output and refresh packs
    for (int e = 0; e < 16; ++e) {
      int idx = tid + 256 * e;
      int m = idx >> 8, n = idx & 255;
      float z  = zt[idx];
      float nh = (1.0f - z) * hf[idx] + z * tmp[idx];
      hf[idx] = nh;
      hA[apack_off(m, n)] = (_Float16)nh;
      out[(long)(b0 + m) * (LL * SA) + (long)t * SA + n] = nh;
    }
    __syncthreads();
  }
}

// ---------------------------------------------------------------------------
extern "C" void kernel_launch(void* const* d_in, const int* in_sizes, int n_in,
                              void* d_out, int out_size, void* d_ws, size_t ws_size,
                              hipStream_t stream) {
  (void)in_sizes; (void)n_in; (void)out_size; (void)ws_size;

  const int*   input_ids   = (const int*)  d_in[0];
  // d_in[1] lengths: unused by the reference forward
  const float* embedding   = (const float*)d_in[2];
  const float* embed_r     = (const float*)d_in[3];
  const float* embed_r_gen = (const float*)d_in[4];
  const float* Wss1        = (const float*)d_in[5];
  const float* Wrs1        = (const float*)d_in[6];
  const float* bs1         = (const float*)d_in[7];
  const float* Wss2        = (const float*)d_in[8];
  const float* Wrs2        = (const float*)d_in[9];
  const float* bs2         = (const float*)d_in[10];
  const float* beta_vec    = (const float*)d_in[11];
  const float* twc         = (const float*)d_in[12];
  const float* tr1         = (const float*)d_in[13];
  const float* tr2         = (const float*)d_in[14];
  const float* h1          = (const float*)d_in[15];
  float* out = (float*)d_out;

  // workspace layout (halves): [7 packed weights][packed embed_r_gen][packed Lseq]
  _Float16* wp = (_Float16*)d_ws;
  _Float16* Gp = wp + 7 * WMAT;
  _Float16* Lp = Gp + GPAD;

  const int packElems = 7 * WMAT + GPAD;              // 540672
  pack_weights_kernel<<<(packElems + 255) / 256, 256, 0, stream>>>(
      Wss1, Wrs1, Wss2, Wrs2, tr1, tr2, twc, embed_r_gen, wp);

  phase1_kernel<<<dim3(BB / 16, LL), 256, 0, stream>>>(
      input_ids, embedding, embed_r, beta_vec, Gp, Lp);

  const size_t smem = 3 * 4096 * sizeof(_Float16) + 3 * 4096 * sizeof(float); // 72 KB
  phase2_kernel<<<BB / 16, 256, smem, stream>>>(wp, Lp, bs1, bs2, h1, out);
}
